// EncoderBlock_3564822855788
// MI455X (gfx1250) — compile-verified, matrix operand order checked by reading, not run on hardware
//
#include <hip/hip_runtime.h>
#include <math.h>

// ---------------- problem constants ----------------
#define NB     4
#define TT     16
#define HH     32
#define WW     32
#define CC     256
#define HIDN   1024
#define LTOK   (NB * TT * HH * WW)   // 65536
#define NHEAD  8
#define HDIM   32

typedef unsigned short u16;
typedef unsigned int   u32;
typedef __attribute__((ext_vector_type(16))) __bf16 bf16x16;
typedef __attribute__((ext_vector_type(8)))  float  f32x8;
typedef __attribute__((ext_vector_type(4)))  u32    u32x4;
typedef __attribute__((ext_vector_type(8)))  int    i32x8;
typedef __attribute__((ext_vector_type(4)))  int    i32x4;

#if defined(__has_builtin)
#if __has_builtin(__builtin_amdgcn_tensor_load_to_lds) && \
    __has_builtin(__builtin_amdgcn_s_wait_tensorcnt)
#define HAVE_TDM 1
#endif
#endif
#ifndef HAVE_TDM
#define HAVE_TDM 0
#endif

union AFrag { bf16x16 v; u32 u[8]; };

__device__ inline u16 f2bf(float f) {
  union { float f; u32 u; } x; x.f = f;
  u32 r = x.u + 0x7FFFu + ((x.u >> 16) & 1u);
  return (u16)(r >> 16);
}
__device__ inline float bf2f(u16 s) {
  union { u32 u; float f; } x; x.u = ((u32)s) << 16; return x.f;
}
__device__ inline float gelu_f(float x) {
  return 0.5f * x * (1.0f + erff(x * 0.7071067811865476f));
}
__device__ inline float wred_sum(float v) {
#pragma unroll
  for (int m = 16; m > 0; m >>= 1) v += __shfl_xor(v, m, 32);
  return v;
}
__device__ inline f32x8 zero8() {
  f32x8 z;
#pragma unroll
  for (int i = 0; i < 8; i++) z[i] = 0.0f;
  return z;
}
// K-packing pattern for 16-bit A/B fragments (ISA 7.12.2):
// lanes 0-15 hold K 0..7 and 16..23, lanes 16-31 hold K 8..15 and 24..31,
// two consecutive K values per VGPR.
__device__ inline int kpat(int j, int hl) {
  return (j < 4) ? (hl * 8 + 2 * j) : (16 + hl * 8 + 2 * (j - 4));
}

#if HAVE_TDM
// Flat shared-aperture addresses keep the LDS byte offset in the low 32 bits.
__device__ inline unsigned lds_off(const void* p) {
  return (unsigned)(unsigned long long)(uintptr_t)p;
}
// Post one 2-D TDM tile: 32 bf16 along K x 128 rows, row stride = stride_elems,
// LDS rows padded 64B data + 16B pad (i.e. the [rows][40] u16 layout).
// D# layout per CDNA5 ISA ch.8 (group0: count/lds/global/type, group1: dims).
__device__ inline void tdm_load_tile(unsigned lds_addr, const u16* tile,
                                     int td0, int td1, int stride_elems) {
  unsigned long long ga = (unsigned long long)(uintptr_t)tile;
  u32x4 g0;
  g0[0] = 1u;                                              // count=1 (valid), user D#
  g0[1] = lds_addr;                                        // lds_addr (bytes)
  g0[2] = (u32)(ga & 0xFFFFFFFFull);                       // global_addr lo
  g0[3] = (u32)((ga >> 32) & 0x01FFFFFFull) | (2u << 30);  // global_addr hi | type=2
  i32x8 g1;
  // data_size=1(2B) | pad_enable | pad_interval=3(16 dwords) | pad_amount=3(4 dwords)
  g1[0] = (1 << 16) | (1 << 20) | (3 << 22) | (3 << 25);
  u32 d0 = (u32)td0, d1 = (u32)td1;
  g1[1] = (int)((d0 & 0xFFFFu) << 16);                     // tensor_dim0[15:0]
  g1[2] = (int)((d0 >> 16) | ((d1 & 0xFFFFu) << 16));      // tensor_dim0[31:16], dim1[15:0]
  g1[3] = (int)((d1 >> 16) | (32u << 16));                 // tensor_dim1[31:16], tile_dim0=32
  g1[4] = 128;                                             // tile_dim1=128, tile_dim2=0
  unsigned long long s0 = (unsigned long long)(u32)stride_elems;
  g1[5] = (int)(s0 & 0xFFFFFFFFull);                       // tensor_dim0_stride lo
  g1[6] = (int)((s0 >> 32) & 0xFFFFull);                   // stride hi, dim1_stride lo=0
  g1[7] = 0;
  i32x4 z4; z4[0] = 0; z4[1] = 0; z4[2] = 0; z4[3] = 0;
#if __clang_major__ >= 23
  i32x8 z8;
#pragma unroll
  for (int i = 0; i < 8; i++) z8[i] = 0;
  __builtin_amdgcn_tensor_load_to_lds(g0, g1, z4, z4, z8, 0);
#else
  __builtin_amdgcn_tensor_load_to_lds(g0, g1, z4, z4, 0);
#endif
}
#endif  // HAVE_TDM

// ---------------- positional embeddings ----------------
__global__ void pos_embed_kernel(float* __restrict__ pos2d, float* __restrict__ pos1d) {
  int i = blockIdx.x * 256 + threadIdx.x;
  const int HWC = HH * WW * CC;
  if (i < HWC) {
    int c = i & (CC - 1);
    int w = (i >> 8) & (WW - 1);
    int h = i >> 13;
    const int halfc = CC / 2;
    float pos; int cc;
    if (c < halfc) { pos = (float)((h & 7) + 1); cc = c; }
    else           { pos = (float)((w & 7) + 1); cc = c - halfc; }
    float f = powf(10000.0f, (float)(2 * (cc >> 1)) / (float)halfc);
    float p = pos / f;
    pos2d[i] = (cc & 1) ? cosf(p) : sinf(p);
  } else {
    int j = i - HWC;                 // j < TT*CC
    int c = j & (CC - 1);
    int t = j >> 8;
    float d = expf(-(float)(2 * (c >> 1)) * logf(10000.0f) / (float)CC);
    float p = (float)t * d;
    pos1d[j] = (c & 1) ? cosf(p) : sinf(p);
  }
}

// ---------------- fp32 -> bf16, transposed (weights: [K][N] -> [N][K]) ----------------
__global__ void cvt_bf16_t_kernel(const float* __restrict__ src, u16* __restrict__ dst,
                                  int K, int N) {
  int i = blockIdx.x * 256 + threadIdx.x;
  if (i < K * N) {
    int k = i / N, n = i - k * N;
    dst[(size_t)n * K + k] = f2bf(src[i]);
  }
}

// ---------------- layernorm family ----------------
// one wave per token (C=256, 8 channels per lane)
// mode 0: spatial  : v_out = bf(ln(x)), qk_out = bf(ln(x + pos2d[h,w]))
// mode 1: single   : v_out = bf(ln(x))
// mode 2: temporal : y = ln(x) -> yout (in place OK); v_out = bf(ln(y));
//                    qk_out = bf(ln(y + pos1d[t]))
__global__ void __launch_bounds__(256) ln_kernel(
    const float* x, const float* __restrict__ pos,
    const float* __restrict__ g, const float* __restrict__ b,
    float* yout, u16* __restrict__ qk_out, u16* __restrict__ v_out, int mode) {
  int tok  = blockIdx.x * 8 + (threadIdx.x >> 5);
  int lane = threadIdx.x & 31;
  size_t base = (size_t)tok * CC;
  float xr[8], gr[8], br[8];
#pragma unroll
  for (int j = 0; j < 8; j++) {
    int c = lane + j * 32;
    xr[j] = x[base + c]; gr[j] = g[c]; br[j] = b[c];
  }
  float s = 0.f, ss = 0.f;
#pragma unroll
  for (int j = 0; j < 8; j++) { s += xr[j]; ss += xr[j] * xr[j]; }
  s = wred_sum(s); ss = wred_sum(ss);
  float m   = s * (1.0f / CC);
  float ist = rsqrtf(ss * (1.0f / CC) - m * m + 1e-5f);

  if (mode == 0) {
    int hw = tok & (HH * WW - 1);
    float pr[8], s2 = 0.f, ss2 = 0.f;
#pragma unroll
    for (int j = 0; j < 8; j++) {
      int c = lane + j * 32;
      pr[j] = xr[j] + pos[(size_t)hw * CC + c];
      s2 += pr[j]; ss2 += pr[j] * pr[j];
    }
    s2 = wred_sum(s2); ss2 = wred_sum(ss2);
    float m2 = s2 * (1.0f / CC);
    float i2 = rsqrtf(ss2 * (1.0f / CC) - m2 * m2 + 1e-5f);
#pragma unroll
    for (int j = 0; j < 8; j++) {
      int c = lane + j * 32;
      v_out[base + c]  = f2bf((xr[j] - m) * ist * gr[j] + br[j]);
      qk_out[base + c] = f2bf((pr[j] - m2) * i2 * gr[j] + br[j]);
    }
  } else if (mode == 1) {
#pragma unroll
    for (int j = 0; j < 8; j++) {
      int c = lane + j * 32;
      v_out[base + c] = f2bf((xr[j] - m) * ist * gr[j] + br[j]);
    }
  } else {
    float yr[8];
#pragma unroll
    for (int j = 0; j < 8; j++) yr[j] = (xr[j] - m) * ist * gr[j] + br[j];
#pragma unroll
    for (int j = 0; j < 8; j++) yout[base + lane + j * 32] = yr[j];
    // v = ln(y)
    float s2 = 0.f, ss2 = 0.f;
#pragma unroll
    for (int j = 0; j < 8; j++) { s2 += yr[j]; ss2 += yr[j] * yr[j]; }
    s2 = wred_sum(s2); ss2 = wred_sum(ss2);
    float m2 = s2 * (1.0f / CC);
    float i2 = rsqrtf(ss2 * (1.0f / CC) - m2 * m2 + 1e-5f);
#pragma unroll
    for (int j = 0; j < 8; j++) {
      int c = lane + j * 32;
      v_out[base + c] = f2bf((yr[j] - m2) * i2 * gr[j] + br[j]);
    }
    // qk = ln(y + pos1d[t])
    int t = (tok >> 10) & (TT - 1);
    float pr[8], s3 = 0.f, ss3 = 0.f;
#pragma unroll
    for (int j = 0; j < 8; j++) {
      int c = lane + j * 32;
      pr[j] = yr[j] + pos[(size_t)t * CC + c];
      s3 += pr[j]; ss3 += pr[j] * pr[j];
    }
    s3 = wred_sum(s3); ss3 = wred_sum(ss3);
    float m3 = s3 * (1.0f / CC);
    float i3 = rsqrtf(ss3 * (1.0f / CC) - m3 * m3 + 1e-5f);
#pragma unroll
    for (int j = 0; j < 8; j++) {
      int c = lane + j * 32;
      qk_out[base + c] = f2bf((pr[j] - m3) * i3 * gr[j] + br[j]);
    }
  }
}

// ---------------- generic bf16 WMMA GEMM ----------------
// C[M,N] = act(A[M,K] @ B[K,N] + bias) (+resid), *alpha.  B given TRANSPOSED [N][K].
// Block tile 128x128, 8 waves (4 x 2), K stepped by 32, double-buffered LDS.
// Tiles staged by the Tensor Data Mover when available (wave 0 posts descriptors,
// s_wait_tensorcnt, waves overlap WMMA with the next tile's DMA).
__global__ void __launch_bounds__(256) gemm_bf16_kernel(
    const u16* __restrict__ A, const u16* __restrict__ Bt, const float* __restrict__ bias,
    const float* resid, float* outF, u16* outB,
    int M, int N, int K, int act, float alpha) {
  __shared__ u16 sA[2][128][40];   // [buf][row][k], rows padded 32+8
  __shared__ u16 sB[2][128][40];   // [buf][col][k]
  const int t    = threadIdx.x;
  const int lane = t & 31;
  const int wave = t >> 5;
  const int wm   = wave >> 1;   // 0..3
  const int wn   = wave & 1;    // 0..1
  const int bm   = blockIdx.x * 128;
  const int bn   = blockIdx.y * 128;
  const int hl   = lane >> 4;
  const int li   = lane & 15;
  const int ksteps = K / 32;

  f32x8 acc[2][4];
#pragma unroll
  for (int i = 0; i < 2; i++)
#pragma unroll
    for (int j = 0; j < 4; j++) acc[i][j] = zero8();

  const u16* Arow = A  + (size_t)bm * K;
  const u16* Brow = Bt + (size_t)bn * K;

#if HAVE_TDM
  if (wave == 0) {   // prologue: post tile 0 into buffer 0
    tdm_load_tile(lds_off(&sA[0][0][0]), Arow, K, 128, K);
    tdm_load_tile(lds_off(&sB[0][0][0]), Brow, K, 128, K);
  }
#endif

  for (int i = 0; i < ksteps; i++) {
    const int b = i & 1;
#if HAVE_TDM
    if (wave == 0) {
      if (i + 1 < ksteps) {   // post next tile, then wait for current (in-order DMA)
        tdm_load_tile(lds_off(&sA[1 - b][0][0]), Arow + (i + 1) * 32, K, 128, K);
        tdm_load_tile(lds_off(&sB[1 - b][0][0]), Brow + (i + 1) * 32, K, 128, K);
        __builtin_amdgcn_s_wait_tensorcnt(2);
      } else {
        __builtin_amdgcn_s_wait_tensorcnt(0);
      }
    }
    __syncthreads();
#else
    {  // synchronous staging fallback: both tiles are plain row-major-K copies
      int r = t >> 1, c = (t & 1) * 16;
      const u16* sa = Arow + (size_t)r * K + i * 32 + c;
      *(u32x4*)&sA[b][r][c]     = *(const u32x4*)sa;
      *(u32x4*)&sA[b][r][c + 8] = *(const u32x4*)(sa + 8);
      const u16* sb = Brow + (size_t)r * K + i * 32 + c;
      *(u32x4*)&sB[b][r][c]     = *(const u32x4*)sb;
      *(u32x4*)&sB[b][r][c + 8] = *(const u32x4*)(sb + 8);
      if (i + 1 < ksteps) {
        __builtin_prefetch(sa + 32, 0, 1);   // global_prefetch_b8
        __builtin_prefetch(sb + 32, 0, 1);
      }
    }
    __syncthreads();
#endif

    AFrag af[2], bfr[4];
#pragma unroll
    for (int ms = 0; ms < 2; ms++) {
      int row = wm * 32 + ms * 16 + li;
#pragma unroll
      for (int j = 0; j < 8; j++)
        af[ms].u[j] = *(const u32*)&sA[b][row][kpat(j, hl)];
    }
#pragma unroll
    for (int ns = 0; ns < 4; ns++) {
      int col = wn * 64 + ns * 16 + li;
#pragma unroll
      for (int j = 0; j < 8; j++)
        bfr[ns].u[j] = *(const u32*)&sB[b][col][kpat(j, hl)];
    }
#pragma unroll
    for (int ms = 0; ms < 2; ms++)
#pragma unroll
      for (int ns = 0; ns < 4; ns++)
        acc[ms][ns] = __builtin_amdgcn_wmma_f32_16x16x32_bf16(
            false, af[ms].v, false, bfr[ns].v, (short)0, acc[ms][ns], false, false);
#if HAVE_TDM
    __syncthreads();   // all waves done with buf b before DMA refills it
#endif
  }

#pragma unroll
  for (int ms = 0; ms < 2; ms++)
#pragma unroll
    for (int ns = 0; ns < 4; ns++)
#pragma unroll
      for (int r = 0; r < 8; r++) {
        int row = bm + wm * 32 + ms * 16 + r + hl * 8;
        int col = bn + wn * 64 + ns * 16 + li;
        float v = acc[ms][ns][r];
        if (bias)  v += bias[col];
        if (act)   v = gelu_f(v);
        if (resid) v += resid[(size_t)row * N + col];
        v *= alpha;
        if (outF) outF[(size_t)row * N + col] = v;
        if (outB) outB[(size_t)row * N + col] = f2bf(v);
      }
}

// ---------------- spatial windowed attention ----------------
// one wave per (window, head); 64 tokens/window, hd=32
__global__ void __launch_bounds__(64) spatial_attn_kernel(
    const u16* __restrict__ Q, const u16* __restrict__ Kp, const u16* __restrict__ V,
    u16* __restrict__ O) {
  __shared__ float sS[2][64][65];
  __shared__ u16   sP[2][64][64];
  __shared__ u16   sVt[2][32][64];
  const int wv   = threadIdx.x >> 5;
  const int lane = threadIdx.x & 31;
  const int hl   = lane >> 4, li = lane & 15;
  const int id   = blockIdx.x * 2 + wv;     // window-head
  const int win  = id >> 3, head = id & 7;
  const int wb = win & 3, hb = (win >> 2) & 3, nt = win >> 4;
  const int cbase = head * HDIM;

  auto tokOf = [&](int q) -> int {
    int py = q >> 3, px = q & 7;
    return nt * (HH * WW) + (hb * 8 + py) * WW + wb * 8 + px;
  };

  // stage V^T (hd-major so K (=key) is contiguous for B fragments)
  for (int idx = lane; idx < 64 * HDIM; idx += 32) {
    int key = idx >> 5, d = idx & 31;
    sVt[wv][d][key] = V[(size_t)tokOf(key) * CC + cbase + d];
  }

  // ---- scores: S = Q @ K^T, 16 WMMAs, operands straight from global ----
  AFrag qf[4], kf[4];
#pragma unroll
  for (int mt = 0; mt < 4; mt++) {
    size_t rb = (size_t)tokOf(mt * 16 + li) * CC + cbase;
#pragma unroll
    for (int j = 0; j < 8; j++) qf[mt].u[j] = *(const u32*)(Q + rb + kpat(j, hl));
  }
#pragma unroll
  for (int nb2 = 0; nb2 < 4; nb2++) {
    size_t rb = (size_t)tokOf(nb2 * 16 + li) * CC + cbase;
#pragma unroll
    for (int j = 0; j < 8; j++) kf[nb2].u[j] = *(const u32*)(Kp + rb + kpat(j, hl));
  }
#pragma unroll
  for (int mt = 0; mt < 4; mt++)
#pragma unroll
    for (int nb2 = 0; nb2 < 4; nb2++) {
      f32x8 s = __builtin_amdgcn_wmma_f32_16x16x32_bf16(
          false, qf[mt].v, false, kf[nb2].v, (short)0, zero8(), false, false);
#pragma unroll
      for (int r = 0; r < 8; r++)
        sS[wv][mt * 16 + r + hl * 8][nb2 * 16 + li] = s[r];
    }
  __syncthreads();

  // ---- softmax over 64 keys (2 rows per lane), scaled by 1/sqrt(32) ----
  const float scale = 0.17677669529663687f;
#pragma unroll
  for (int rr = 0; rr < 2; rr++) {
    int r = lane * 2 + rr;
    float mx = -1e30f;
    for (int k = 0; k < 64; k++) mx = fmaxf(mx, sS[wv][r][k]);
    float sum = 0.f;
    for (int k = 0; k < 64; k++) sum += expf((sS[wv][r][k] - mx) * scale);
    float inv = 1.0f / sum;
    for (int k = 0; k < 64; k++)
      sP[wv][r][k] = f2bf(expf((sS[wv][r][k] - mx) * scale) * inv);
  }
  __syncthreads();

  // ---- out = P @ V : 16 WMMAs ----
  f32x8 acc[4][2];
#pragma unroll
  for (int i = 0; i < 4; i++)
#pragma unroll
    for (int j = 0; j < 2; j++) acc[i][j] = zero8();
#pragma unroll
  for (int kst = 0; kst < 2; kst++) {
    AFrag pf[4], vf[2];
#pragma unroll
    for (int mt = 0; mt < 4; mt++)
#pragma unroll
      for (int j = 0; j < 8; j++)
        pf[mt].u[j] = *(const u32*)&sP[wv][mt * 16 + li][kst * 32 + kpat(j, hl)];
#pragma unroll
    for (int nd = 0; nd < 2; nd++)
#pragma unroll
      for (int j = 0; j < 8; j++)
        vf[nd].u[j] = *(const u32*)&sVt[wv][nd * 16 + li][kst * 32 + kpat(j, hl)];
#pragma unroll
    for (int mt = 0; mt < 4; mt++)
#pragma unroll
      for (int nd = 0; nd < 2; nd++)
        acc[mt][nd] = __builtin_amdgcn_wmma_f32_16x16x32_bf16(
            false, pf[mt].v, false, vf[nd].v, (short)0, acc[mt][nd], false, false);
  }
#pragma unroll
  for (int mt = 0; mt < 4; mt++)
#pragma unroll
    for (int nd = 0; nd < 2; nd++)
#pragma unroll
      for (int r = 0; r < 8; r++) {
        int q = mt * 16 + r + hl * 8;
        O[(size_t)tokOf(q) * CC + cbase + nd * 16 + li] = f2bf(acc[mt][nd][r]);
      }
}

// ---------------- temporal attention over T=16 ----------------
// one wave per ((n,h,w), head); softmax lives in registers (16-lane shfl)
__global__ void __launch_bounds__(256) temporal_attn_kernel(
    const u16* __restrict__ Q, const u16* __restrict__ Kp, const u16* __restrict__ V,
    u16* __restrict__ O) {
  __shared__ u16 sP[8][16][32];
  __shared__ u16 sVt[8][32][32];
  const int wv   = threadIdx.x >> 5;
  const int lane = threadIdx.x & 31;
  const int hl   = lane >> 4, li = lane & 15;
  const int id   = blockIdx.x * 8 + wv;
  const int bidx = id >> 3, head = id & 7;
  const int n = bidx >> 10, hw = bidx & 1023;
  const int cbase = head * HDIM;
  auto tokOf = [&](int t) -> int { return n * (TT * HH * WW) + t * (HH * WW) + hw; };

  // stage V^T, zero-padded K 16 -> 32 (avoid NaN from 0*garbage)
  for (int idx = lane; idx < 32 * 32; idx += 32) {
    int d = idx >> 5, key = idx & 31;
    sVt[wv][d][key] = (key < TT) ? V[(size_t)tokOf(key) * CC + cbase + d] : (u16)0;
  }

  AFrag qf, kf;
  size_t rowb = (size_t)tokOf(li) * CC + cbase;
#pragma unroll
  for (int j = 0; j < 8; j++) qf.u[j] = *(const u32*)(Q  + rowb + kpat(j, hl));
#pragma unroll
  for (int j = 0; j < 8; j++) kf.u[j] = *(const u32*)(Kp + rowb + kpat(j, hl));

  f32x8 s = __builtin_amdgcn_wmma_f32_16x16x32_bf16(
      false, qf.v, false, kf.v, (short)0, zero8(), false, false);

  const float scale = 0.17677669529663687f;
#pragma unroll
  for (int r = 0; r < 8; r++) {
    float v = s[r] * scale;
    float mx = v;
#pragma unroll
    for (int m = 8; m > 0; m >>= 1) mx = fmaxf(mx, __shfl_xor(mx, m, 32));
    float e = expf(v - mx);
    float sum = e;
#pragma unroll
    for (int m = 8; m > 0; m >>= 1) sum += __shfl_xor(sum, m, 32);
    sP[wv][r + hl * 8][li]      = f2bf(e / sum);
    sP[wv][r + hl * 8][li + 16] = 0;   // zero-pad K
  }
  __syncthreads();

  AFrag pf, vf0, vf1;
#pragma unroll
  for (int j = 0; j < 8; j++) pf.u[j]  = *(const u32*)&sP[wv][li][kpat(j, hl)];
#pragma unroll
  for (int j = 0; j < 8; j++) vf0.u[j] = *(const u32*)&sVt[wv][li][kpat(j, hl)];
#pragma unroll
  for (int j = 0; j < 8; j++) vf1.u[j] = *(const u32*)&sVt[wv][16 + li][kpat(j, hl)];
  f32x8 o0 = __builtin_amdgcn_wmma_f32_16x16x32_bf16(
      false, pf.v, false, vf0.v, (short)0, zero8(), false, false);
  f32x8 o1 = __builtin_amdgcn_wmma_f32_16x16x32_bf16(
      false, pf.v, false, vf1.v, (short)0, zero8(), false, false);
#pragma unroll
  for (int r = 0; r < 8; r++) {
    size_t ob = (size_t)tokOf(r + hl * 8) * CC + cbase;
    O[ob + li]      = f2bf(o0[r]);
    O[ob + 16 + li] = f2bf(o1[r]);
  }
}

// ---------------- depthwise 3x3 + GELU (bf16 in/out, fp32 math) ----------------
__global__ void __launch_bounds__(256) dwconv_gelu_kernel(
    const u16* __restrict__ hin, const float* __restrict__ wk, const float* __restrict__ wb,
    u16* __restrict__ hout) {
  size_t i = (size_t)blockIdx.x * 256 + threadIdx.x;
  int c   = (int)(i & (HIDN - 1));
  int sp  = (int)(i >> 10);
  int w   = sp & (WW - 1);
  int h   = (sp >> 5) & (HH - 1);
  int ntb = sp >> 10;
  float acc = 0.f;
#pragma unroll
  for (int dy = -1; dy <= 1; dy++) {
    int y = h + dy;
    if (y < 0 || y >= HH) continue;
#pragma unroll
    for (int dx = -1; dx <= 1; dx++) {
      int x2 = w + dx;
      if (x2 < 0 || x2 >= WW) continue;
      size_t idx = (((size_t)(ntb * HH + y) * WW + x2) << 10) + c;
      acc += bf2f(hin[idx]) * wk[c * 9 + (dy + 1) * 3 + (dx + 1)];
    }
  }
  acc += wb[c];
  hout[i] = f2bf(gelu_f(acc));
}

// ---------------- host orchestration ----------------
static void launch_gemm(const u16* A, const u16* Bt, const float* bias, const float* resid,
                        float* outF, u16* outB, int M, int N, int K, int act, float alpha,
                        hipStream_t s) {
  dim3 grid(M / 128, N / 128);
  gemm_bf16_kernel<<<grid, 256, 0, s>>>(A, Bt, bias, resid, outF, outB, M, N, K, act, alpha);
}

extern "C" void kernel_launch(void* const* d_in, const int* in_sizes, int n_in,
                              void* d_out, int out_size, void* d_ws, size_t ws_size,
                              hipStream_t stream) {
  (void)in_sizes; (void)n_in; (void)out_size; (void)ws_size;
  const float* x   = (const float*)d_in[0];
  const float* g1  = (const float*)d_in[1];  const float* b1  = (const float*)d_in[2];
  const float* g2  = (const float*)d_in[3];  const float* b2  = (const float*)d_in[4];
  const float* g3  = (const float*)d_in[5];  const float* b3  = (const float*)d_in[6];
  const float* g4  = (const float*)d_in[7];  const float* b4  = (const float*)d_in[8];
  const float* sWq = (const float*)d_in[9];  const float* sbq = (const float*)d_in[10];
  const float* sWk = (const float*)d_in[11]; const float* sbk = (const float*)d_in[12];
  const float* sWv = (const float*)d_in[13]; const float* sbv = (const float*)d_in[14];
  const float* sWo = (const float*)d_in[15]; const float* sbo = (const float*)d_in[16];
  const float* tWq = (const float*)d_in[17]; const float* tbq = (const float*)d_in[18];
  const float* tWk = (const float*)d_in[19]; const float* tbk = (const float*)d_in[20];
  const float* tWv = (const float*)d_in[21]; const float* tbv = (const float*)d_in[22];
  const float* tWo = (const float*)d_in[23]; const float* tbo = (const float*)d_in[24];
  const float* cW1 = (const float*)d_in[25]; const float* cb1 = (const float*)d_in[26];
  const float* dwk = (const float*)d_in[27]; const float* dwb = (const float*)d_in[28];
  const float* cW2 = (const float*)d_in[29]; const float* cb2 = (const float*)d_in[30];
  const float* fW1 = (const float*)d_in[31]; const float* fb1 = (const float*)d_in[32];
  const float* fW2 = (const float*)d_in[33]; const float* fb2 = (const float*)d_in[34];

  char* ws = (char*)d_ws;
  size_t off = 0;
  auto take = [&](size_t bytes) -> char* {
    char* p = ws + off; off += (bytes + 255) & ~(size_t)255; return p;
  };
  float* pos2d = (float*)take((size_t)HH * WW * CC * 4);
  float* pos1d = (float*)take((size_t)TT * CC * 4);
  u16* wsq = (u16*)take((size_t)CC * CC * 2);      // all weight copies are [N][K] bf16
  u16* wsk = (u16*)take((size_t)CC * CC * 2);
  u16* wsv = (u16*)take((size_t)CC * CC * 2);
  u16* wso = (u16*)take((size_t)CC * CC * 2);
  u16* wtq = (u16*)take((size_t)CC * CC * 2);
  u16* wtk = (u16*)take((size_t)CC * CC * 2);
  u16* wtv = (u16*)take((size_t)CC * CC * 2);
  u16* wto = (u16*)take((size_t)CC * CC * 2);
  u16* wc1 = (u16*)take((size_t)CC * HIDN * 2);
  u16* wc2 = (u16*)take((size_t)HIDN * CC * 2);
  u16* wf1 = (u16*)take((size_t)CC * HIDN * 2);
  u16* wf2 = (u16*)take((size_t)HIDN * CC * 2);
  float* xbuf = (float*)take((size_t)LTOK * CC * 4);
  u16* lnA = (u16*)take((size_t)LTOK * CC * 2);   // qk operand
  u16* lnB = (u16*)take((size_t)LTOK * CC * 2);   // v operand
  u16* qb  = (u16*)take((size_t)LTOK * CC * 2);
  u16* kb  = (u16*)take((size_t)LTOK * CC * 2);
  u16* vb  = (u16*)take((size_t)LTOK * CC * 2);
  u16* ab  = (u16*)take((size_t)LTOK * CC * 2);
  u16* h1  = (u16*)take((size_t)LTOK * HIDN * 2); // fits in 192MB L2 at bf16
  u16* h2  = (u16*)take((size_t)LTOK * HIDN * 2);

  auto cvtT = [&](const float* s, u16* d, int Kd, int Nd) {
    cvt_bf16_t_kernel<<<(Kd * Nd + 255) / 256, 256, 0, stream>>>(s, d, Kd, Nd);
  };

  // positional embeddings + transposed bf16 weights
  pos_embed_kernel<<<(HH * WW * CC + TT * CC) / 256, 256, 0, stream>>>(pos2d, pos1d);
  cvtT(sWq, wsq, CC, CC); cvtT(sWk, wsk, CC, CC); cvtT(sWv, wsv, CC, CC); cvtT(sWo, wso, CC, CC);
  cvtT(tWq, wtq, CC, CC); cvtT(tWk, wtk, CC, CC); cvtT(tWv, wtv, CC, CC); cvtT(tWo, wto, CC, CC);
  cvtT(cW1, wc1, CC, HIDN); cvtT(cW2, wc2, HIDN, CC);
  cvtT(fW1, wf1, CC, HIDN); cvtT(fW2, wf2, HIDN, CC);

  const int LNB = LTOK / 8;      // 8192 blocks of 8 tokens
  // ---- stage 1: spatial windowed MHSA ----
  ln_kernel<<<LNB, 256, 0, stream>>>(x, pos2d, g1, b1, nullptr, lnA, lnB, 0);
  launch_gemm(lnA, wsq, sbq, nullptr, nullptr, qb, LTOK, CC, CC, 0, 1.f, stream);
  launch_gemm(lnA, wsk, sbk, nullptr, nullptr, kb, LTOK, CC, CC, 0, 1.f, stream);
  launch_gemm(lnB, wsv, sbv, nullptr, nullptr, vb, LTOK, CC, CC, 0, 1.f, stream);
  spatial_attn_kernel<<<(LTOK / 64) * NHEAD / 2, 64, 0, stream>>>(qb, kb, vb, ab);
  launch_gemm(ab, wso, sbo, x, xbuf, nullptr, LTOK, CC, CC, 0, 1.f, stream);

  // ---- stage 2: conv FFN ----
  ln_kernel<<<LNB, 256, 0, stream>>>(xbuf, nullptr, g2, b2, nullptr, nullptr, lnA, 1);
  launch_gemm(lnA, wc1, cb1, nullptr, nullptr, h1, LTOK, HIDN, CC, 0, 1.f, stream);
  dwconv_gelu_kernel<<<(size_t)LTOK * HIDN / 256, 256, 0, stream>>>(h1, dwk, dwb, h2);
  launch_gemm(h2, wc2, cb2, xbuf, xbuf, nullptr, LTOK, CC, HIDN, 0, 1.f, stream);

  // ---- stage 3: temporal MHSA ----
  ln_kernel<<<LNB, 256, 0, stream>>>(xbuf, pos1d, g3, b3, xbuf, lnA, lnB, 2);
  launch_gemm(lnA, wtq, tbq, nullptr, nullptr, qb, LTOK, CC, CC, 0, 1.f, stream);
  launch_gemm(lnA, wtk, tbk, nullptr, nullptr, kb, LTOK, CC, CC, 0, 1.f, stream);
  launch_gemm(lnB, wtv, tbv, nullptr, nullptr, vb, LTOK, CC, CC, 0, 1.f, stream);
  temporal_attn_kernel<<<(NB * HH * WW) * NHEAD / 8, 256, 0, stream>>>(qb, kb, vb, ab);
  launch_gemm(ab, wto, tbo, xbuf, xbuf, nullptr, LTOK, CC, CC, 0, 1.f, stream);

  // ---- stage 4: FFN, output doubled (x = x + x), no residual ----
  ln_kernel<<<LNB, 256, 0, stream>>>(xbuf, nullptr, g4, b4, nullptr, nullptr, lnA, 1);
  launch_gemm(lnA, wf1, fb1, nullptr, nullptr, h1, LTOK, HIDN, CC, 1 /*gelu*/, 1.f, stream);
  launch_gemm(h1, wf2, fb2, nullptr, (float*)d_out, nullptr, LTOK, CC, HIDN, 0, 2.0f, stream);
}
